// Cifar10ConvBNN_39187281609013
// MI455X (gfx1250) — matrix operand order repaired and checked by reading, not compiled, not run on hardware
//
#include <hip/hip_runtime.h>
#include <hip/hip_bf16.h>
#include <cstdint>

typedef __attribute__((ext_vector_type(16))) _Float16 v16h;
typedef __attribute__((ext_vector_type(8)))  _Float16 v8h;
typedef __attribute__((ext_vector_type(8)))  float    v8f;
typedef int v4i __attribute__((vector_size(16)));   // matches async-LDS builtin param

#define EPSV 1e-5f

// ---------------------------------------------------------------------------
// CDNA5 async global->LDS copy (ASYNCcnt-tracked), with portable fallback.
// ---------------------------------------------------------------------------
#if defined(__gfx1250__) && __has_builtin(__builtin_amdgcn_global_load_async_to_lds_b128)
#define USE_ASYNC_LDS 1
#else
#define USE_ASYNC_LDS 0
#endif

__device__ __forceinline__ void async_copy_b128(const _Float16* gsrc, _Float16* ldst) {
#if USE_ASYNC_LDS
    __builtin_amdgcn_global_load_async_to_lds_b128(
        (__attribute__((address_space(1))) v4i*)(uintptr_t)gsrc,
        (__attribute__((address_space(3))) v4i*)(unsigned)(uintptr_t)ldst,
        0, 0);
#else
    *(v8h*)ldst = *(const v8h*)gsrc;
#endif
}

__device__ __forceinline__ void wait_async0() {
#if USE_ASYNC_LDS
#if __has_builtin(__builtin_amdgcn_s_wait_asynccnt)
    __builtin_amdgcn_s_wait_asynccnt(0);
#else
    asm volatile("s_wait_asynccnt 0x0" ::: "memory");
#endif
#endif
}

// ---------------------------------------------------------------------------
// Binarize weights (sign -> +/-1 f16) and pack into WMMA-ready LDS tile
// layout: tile (kb, nb) holds 128 rows (n) of 32 k-values, 64B per row:
//   out[(((kb*NB + nb)*128 + n)*32 + k]  ; zero-filled outside (K, O).
// ---------------------------------------------------------------------------
__global__ void binarize_pack(const float* __restrict__ w, _Float16* __restrict__ out,
                              int O, int K, int NB, long total) {
    long idx = (long)blockIdx.x * blockDim.x + threadIdx.x;
    if (idx < total) {
        int k  = (int)(idx & 31);
        int n  = (int)((idx >> 5) & 127);
        long t = idx >> 12;
        int nb = (int)(t % NB);
        int kb = (int)(t / NB);
        int gk = kb * 32 + k;
        int gn = nb * 128 + n;
        _Float16 v = (_Float16)0.f;
        if (gk < K && gn < O)
            v = (_Float16)(w[(long)gn * K + gk] >= 0.f ? 1.f : -1.f);
        out[idx] = v;
    }
}

// ---------------------------------------------------------------------------
// Implicit-GEMM 3x3 conv, replication padding, binarized f16 weights, WMMA.
// Block tile 128(M) x 128(N), 8 wave32s; each wave owns 32x64 = 2x4 WMMA tiles.
// Double-buffered LDS; weights staged via async global->LDS b128.
// Requires: N*H*W % 128 == 0, Cout % 128 == 0 (true for all layers).
// ---------------------------------------------------------------------------
__global__ __launch_bounds__(256)
void conv3x3_bnn_wmma(const float* __restrict__ x, const _Float16* __restrict__ wpk,
                      const float* __restrict__ bias, float* __restrict__ y,
                      int N, int Cin, int H, int W, int Cout, int NB) {
    const int HW   = H * W;
    const int Ktot = Cin * 9;
    __shared__ __align__(16) _Float16 As[2][128][40];  // [buf][m][k]
    __shared__ __align__(16) _Float16 Bs[2][128][40];  // [buf][n][k]

    const int tid  = threadIdx.x;
    const int lane = tid & 31;
    const int wv   = tid >> 5;
    const int wm   = wv & 3;            // 4 M groups of 32 rows
    const int wn   = wv >> 2;           // 2 N groups of 64 cols
    const int m0   = blockIdx.x * 128;
    const int n0   = blockIdx.y * 128;

    const int half = lane >> 4;
    const int kbA  = half << 3;
    const int kbB  = half << 3;
    const int lm   = lane & 15;

    // ---- hoisted im2col row decode (k-independent) ----
    const int rbaseA = tid >> 5;
    const int colA   = tid & 31;
    int boff[16]; int hps[16]; int wps[16];
#pragma unroll
    for (int i = 0; i < 16; ++i) {
        int gm = m0 + rbaseA + i * 8;
        int wp = gm % W; int t = gm / W;
        int hp = t % H;  int ni = t / H;
        boff[i] = ni * (Cin * HW);
        hps[i]  = hp; wps[i] = wp;
    }
    // ---- incremental k decode state (advances by 32 per stage) ----
    int kcur = colA;
    int ci0  = colA / 9;
    int rs   = colA - ci0 * 9;
    int coff = ci0 * HW;

    v8f acc[2][4] = {};

    auto stageA = [&](int bufs) {
        bool kok = kcur < Ktot;
        int r = (rs >= 6) ? 2 : (rs >= 3 ? 1 : 0);
        int s = rs - r * 3;
#pragma unroll
        for (int i = 0; i < 16; ++i) {
            float v = 0.f;
            if (kok) {
                int ih = hps[i] + r - 1; ih = ih < 0 ? 0 : (ih > H - 1 ? H - 1 : ih);
                int iw = wps[i] + s - 1; iw = iw < 0 ? 0 : (iw > W - 1 ? W - 1 : iw);
                v = x[(long)(boff[i] + coff + ih * W + iw)];
            }
            As[bufs][rbaseA + i * 8][colA] = (_Float16)v;
        }
        kcur += 32; rs += 5; coff += 3 * HW;          // advance gk by 32 = 3*9 + 5
        if (rs >= 9) { rs -= 9; coff += HW; }
    };

    auto stageB = [&](int k0s, int bufs) {
        const _Float16* base = wpk + ((long)((k0s >> 5) * NB + blockIdx.y) * 128) * 32;
#pragma unroll
        for (int p = 0; p < 2; ++p) {
            int ch = tid + p * 256;            // 0..511
            int n = ch >> 2, q = ch & 3;
            async_copy_b128(base + n * 32 + q * 8, &Bs[bufs][n][q * 8]);
        }
    };

    auto compute = [&](int bufs) {
        v16h af[2], bf[4];
#pragma unroll
        for (int mi = 0; mi < 2; ++mi) {
            const _Float16* p = &As[bufs][wm * 32 + mi * 16 + lm][0];
            v8h lo = *(const v8h*)(p + kbA);
            v8h hi = *(const v8h*)(p + 16 + kbA);
            af[mi] = __builtin_shufflevector(lo, hi, 0,1,2,3,4,5,6,7,8,9,10,11,12,13,14,15);
        }
#pragma unroll
        for (int nj = 0; nj < 4; ++nj) {
            const _Float16* p = &Bs[bufs][wn * 64 + nj * 16 + lm][0];
            v8h lo = *(const v8h*)(p + kbB);
            v8h hi = *(const v8h*)(p + 16 + kbB);
            bf[nj] = __builtin_shufflevector(lo, hi, 0,1,2,3,4,5,6,7,8,9,10,11,12,13,14,15);
        }
#pragma unroll
        for (int mi = 0; mi < 2; ++mi)
#pragma unroll
            for (int nj = 0; nj < 4; ++nj)
                acc[mi][nj] = __builtin_amdgcn_wmma_f32_16x16x32_f16(
                    false, af[mi], false, bf[nj], (short)0, acc[mi][nj], false, false);
    };

    stageA(0);
    stageB(0, 0);
    for (int k0 = 0; k0 < Ktot; k0 += 32) {
        int buf = (k0 >> 5) & 1;
        wait_async0();
        __syncthreads();
        if (k0 + 32 < Ktot) {
            stageA(buf ^ 1);
            stageB(k0 + 32, buf ^ 1);
        }
        compute(buf);
    }

    // ---- epilogue: scatter D (NCHW) + bias ----
    float bz[4];
#pragma unroll
    for (int nj = 0; nj < 4; ++nj) bz[nj] = bias[n0 + wn * 64 + nj * 16 + lm];
#pragma unroll
    for (int mi = 0; mi < 2; ++mi) {
#pragma unroll
        for (int r = 0; r < 8; ++r) {
            int mg = m0 + wm * 32 + mi * 16 + r + (half << 3);
            int wp = mg % W; int t = mg / W;
            int hp = t % H;  int ni = t / H;
            long sp = (long)hp * W + wp;
#pragma unroll
            for (int nj = 0; nj < 4; ++nj) {
                int cout = n0 + wn * 64 + nj * 16 + lm;
                y[((long)(ni * Cout + cout)) * HW + sp] = acc[mi][nj][r] + bz[nj];
            }
        }
    }
}

// ---------------------------------------------------------------------------
// Dense GEMM (FC): y[M,Nout] = x[M,K]*wpk + bias.  M%128==0, K%32==0.
// ---------------------------------------------------------------------------
__global__ __launch_bounds__(256)
void gemm_bnn_wmma(const float* __restrict__ x, const _Float16* __restrict__ wpk,
                   const float* __restrict__ bias, float* __restrict__ y,
                   int M, int K, int Nout, int NB) {
    __shared__ __align__(16) _Float16 As[2][128][40];
    __shared__ __align__(16) _Float16 Bs[2][128][40];
    typedef __attribute__((ext_vector_type(4))) _Float16 h4;
    const int tid  = threadIdx.x;
    const int lane = tid & 31;
    const int wv   = tid >> 5;
    const int wm   = wv & 3, wn = wv >> 2;
    const int m0   = blockIdx.x * 128, n0 = blockIdx.y * 128;
    const int half = lane >> 4;
    const int kbA  = half << 3;
    const int kbB  = half << 3;
    const int lm   = lane & 15;

    const int rowA  = tid >> 3;
    const int col4A = (tid & 7) * 4;

    v8f acc[2][4] = {};

    auto stageA = [&](int k0s, int bufs) {
#pragma unroll
        for (int p = 0; p < 4; ++p) {
            int row = rowA + p * 32;
            const float4 f = *(const float4*)&x[(long)(m0 + row) * K + k0s + col4A];
            h4 hv;
            hv[0] = (_Float16)f.x; hv[1] = (_Float16)f.y;
            hv[2] = (_Float16)f.z; hv[3] = (_Float16)f.w;
            *(h4*)&As[bufs][row][col4A] = hv;
        }
    };

    auto stageB = [&](int k0s, int bufs) {
        const _Float16* base = wpk + ((long)((k0s >> 5) * NB + blockIdx.y) * 128) * 32;
#pragma unroll
        for (int p = 0; p < 2; ++p) {
            int ch = tid + p * 256;
            int n = ch >> 2, q = ch & 3;
            async_copy_b128(base + n * 32 + q * 8, &Bs[bufs][n][q * 8]);
        }
    };

    auto compute = [&](int bufs) {
        v16h af[2], bf[4];
#pragma unroll
        for (int mi = 0; mi < 2; ++mi) {
            const _Float16* p = &As[bufs][wm * 32 + mi * 16 + lm][0];
            v8h lo = *(const v8h*)(p + kbA);
            v8h hi = *(const v8h*)(p + 16 + kbA);
            af[mi] = __builtin_shufflevector(lo, hi, 0,1,2,3,4,5,6,7,8,9,10,11,12,13,14,15);
        }
#pragma unroll
        for (int nj = 0; nj < 4; ++nj) {
            const _Float16* p = &Bs[bufs][wn * 64 + nj * 16 + lm][0];
            v8h lo = *(const v8h*)(p + kbB);
            v8h hi = *(const v8h*)(p + 16 + kbB);
            bf[nj] = __builtin_shufflevector(lo, hi, 0,1,2,3,4,5,6,7,8,9,10,11,12,13,14,15);
        }
#pragma unroll
        for (int mi = 0; mi < 2; ++mi)
#pragma unroll
            for (int nj = 0; nj < 4; ++nj)
                acc[mi][nj] = __builtin_amdgcn_wmma_f32_16x16x32_f16(
                    false, af[mi], false, bf[nj], (short)0, acc[mi][nj], false, false);
    };

    stageA(0, 0);
    stageB(0, 0);
    for (int k0 = 0; k0 < K; k0 += 32) {
        int buf = (k0 >> 5) & 1;
        wait_async0();
        __syncthreads();
        if (k0 + 32 < K) {
            stageA(k0 + 32, buf ^ 1);
            stageB(k0 + 32, buf ^ 1);
        }
        compute(buf);
    }

#pragma unroll
    for (int nj = 0; nj < 4; ++nj) {
        int cout = n0 + wn * 64 + nj * 16 + lm;
        if (cout < Nout) {
            float bzv = bias[cout];
#pragma unroll
            for (int mi = 0; mi < 2; ++mi)
#pragma unroll
                for (int r = 0; r < 8; ++r) {
                    int mg = m0 + wm * 32 + mi * 16 + r + (half << 3);
                    y[(long)mg * Nout + cout] = acc[mi][nj][r] + bzv;
                }
        }
    }
}

// ---------------------------------------------------------------------------
// 2x2 max pool (stride 2), NCHW.
// ---------------------------------------------------------------------------
__global__ void maxpool2(const float* __restrict__ in, float* __restrict__ out,
                         int N, int C, int H, int W) {
    int H2 = H >> 1, W2 = W >> 1;
    long total = (long)N * C * H2 * W2;
    long idx = (long)blockIdx.x * blockDim.x + threadIdx.x;
    if (idx < total) {
        int w2 = (int)(idx % W2); long t = idx / W2;
        int h2 = (int)(t % H2);   t /= H2;
        int c  = (int)(t % C);
        int n  = (int)(t / C);
        const float* p = in + ((long)(n * C + c) * H + h2 * 2) * W + w2 * 2;
        out[idx] = fmaxf(fmaxf(p[0], p[1]), fmaxf(p[W], p[W + 1]));
    }
}

// ---------------------------------------------------------------------------
// Per-channel batch stats -> fused BN scale/shift. One block per channel.
// ---------------------------------------------------------------------------
__global__ __launch_bounds__(256)
void channel_stats(const float* __restrict__ x, const float* __restrict__ g,
                   const float* __restrict__ bt, float* __restrict__ scale,
                   float* __restrict__ shift, int N, int C, int HW) {
    const int c = blockIdx.x;
    const long cnt = (long)N * HW;
    float s = 0.f, q = 0.f;
    for (long i = threadIdx.x; i < cnt; i += 256) {
        long n = i / HW, hw = i - n * HW;
        float v = x[((long)n * C + c) * HW + hw];
        s += v; q += v * v;
    }
    __shared__ float ss[256], sq[256];
    ss[threadIdx.x] = s; sq[threadIdx.x] = q;
    __syncthreads();
    for (int st = 128; st > 0; st >>= 1) {
        if ((int)threadIdx.x < st) {
            ss[threadIdx.x] += ss[threadIdx.x + st];
            sq[threadIdx.x] += sq[threadIdx.x + st];
        }
        __syncthreads();
    }
    if (threadIdx.x == 0) {
        float mean = ss[0] / (float)cnt;
        float var  = sq[0] / (float)cnt - mean * mean;
        var = var < 0.f ? 0.f : var;
        float sc = g[c] * rsqrtf(var + EPSV);
        scale[c] = sc;
        shift[c] = bt[c] - mean * sc;
    }
}

// ---------------------------------------------------------------------------
// Apply BN (+ optional hardtanh). in/out may alias.
// ---------------------------------------------------------------------------
__global__ void bn_apply(const float* __restrict__ in, float* __restrict__ out,
                         const float* __restrict__ scale, const float* __restrict__ shift,
                         long total, int C, int HW, int do_clip) {
    long idx = (long)blockIdx.x * blockDim.x + threadIdx.x;
    if (idx < total) {
        int c = (int)((idx / HW) % C);
        float v = scale[c] * in[idx] + shift[c];
        if (do_clip) v = fminf(1.f, fmaxf(-1.f, v));
        out[idx] = v;
    }
}

// ---------------------------------------------------------------------------
// Orchestration
// ---------------------------------------------------------------------------
extern "C" void kernel_launch(void* const* d_in, const int* in_sizes, int n_in,
                              void* d_out, int out_size, void* d_ws, size_t ws_size,
                              hipStream_t stream) {
    (void)in_sizes; (void)n_in; (void)out_size; (void)ws_size;
    const int N = 256;

    char* ws = (char*)d_ws;
    float*    bufA  = (float*)ws;                                   // 134,217,728 B
    float*    bufB  = (float*)(ws + (size_t)134217728);             // 134,217,728 B
    _Float16* wbuf  = (_Float16*)(ws + (size_t)268435456);          //  25,165,824 B
    float*    scale = (float*)(ws + (size_t)268435456 + 25165824);
    float*    shift = scale + 2048;

    struct LCfg { int Cin, Cout, H, W, pool; };
    const LCfg cfg[6] = {
        {  3, 128, 32, 32, 0}, {128, 128, 32, 32, 1},
        {128, 256, 16, 16, 0}, {256, 256, 16, 16, 1},
        {256, 512,  8,  8, 0}, {512, 512,  8,  8, 1}
    };

    const float* cur = (const float*)d_in[0];

    for (int L = 0; L < 6; ++L) {
        const LCfg& c = cfg[L];
        const float* cw = (const float*)d_in[1 + L * 4 + 0];
        const float* cb = (const float*)d_in[1 + L * 4 + 1];
        const float* g  = (const float*)d_in[1 + L * 4 + 2];
        const float* bt = (const float*)d_in[1 + L * 4 + 3];

        const int Ktot = c.Cin * 9;
        const int KB = (Ktot + 31) / 32;
        const int NB = (c.Cout + 127) / 128;
        long ptotal = (long)KB * 32 * NB * 128;
        binarize_pack<<<(unsigned)((ptotal + 255) / 256), 256, 0, stream>>>(
            cw, wbuf, c.Cout, Ktot, NB, ptotal);

        float* dst = (cur == bufA) ? bufB : bufA;
        long Mtot = (long)N * c.H * c.W;
        dim3 grid((unsigned)(Mtot / 128), (unsigned)NB);
        conv3x3_bnn_wmma<<<grid, 256, 0, stream>>>(cur, wbuf, cb, dst,
                                                   N, c.Cin, c.H, c.W, c.Cout, NB);
        cur = dst;

        int Hc = c.H, Wc = c.W;
        if (c.pool) {
            float* pd = (cur == bufA) ? bufB : bufA;
            Hc >>= 1; Wc >>= 1;
            long pt = (long)N * c.Cout * Hc * Wc;
            maxpool2<<<(unsigned)((pt + 255) / 256), 256, 0, stream>>>(cur, pd, N, c.Cout, c.H, c.W);
            cur = pd;
        }
        const int HW = Hc * Wc;
        channel_stats<<<c.Cout, 256, 0, stream>>>(cur, g, bt, scale, shift, N, c.Cout, HW);
        long tot = (long)N * c.Cout * HW;
        bn_apply<<<(unsigned)((tot + 255) / 256), 256, 0, stream>>>(
            cur, (float*)cur, scale, shift, tot, c.Cout, HW, 1);
    }

    // FC layers: conv6 output (256 x 512 x 4 x 4) flattens NCHW-contiguously to 256 x 8192
    const int fdims[3][2] = {{8192, 1536}, {1536, 1536}, {1536, 10}};
    for (int l = 0; l < 3; ++l) {
        const float* w  = (const float*)d_in[25 + l * 4 + 0];
        const float* b  = (const float*)d_in[25 + l * 4 + 1];
        const float* gl = (const float*)d_in[25 + l * 4 + 2];
        const float* bl = (const float*)d_in[25 + l * 4 + 3];
        const int K = fdims[l][0], F = fdims[l][1];
        const int KB = K / 32;
        const int NB = (F + 127) / 128;
        long ptotal = (long)KB * 32 * NB * 128;
        binarize_pack<<<(unsigned)((ptotal + 255) / 256), 256, 0, stream>>>(
            w, wbuf, F, K, NB, ptotal);

        float* dst = (cur == bufA) ? bufB : bufA;
        dim3 grid((unsigned)(N / 128), (unsigned)NB);
        gemm_bnn_wmma<<<grid, 256, 0, stream>>>(cur, wbuf, b, dst, N, K, F, NB);
        cur = dst;

        channel_stats<<<F, 256, 0, stream>>>(cur, gl, bl, scale, shift, N, F, 1);
        long tot = (long)N * F;
        float* outp = (l == 2) ? (float*)d_out : (float*)cur;
        bn_apply<<<(unsigned)((tot + 255) / 256), 256, 0, stream>>>(
            cur, outp, scale, shift, tot, F, 1, (l < 2) ? 1 : 0);
    }
}
